// MultiheadAttention_42116449304751
// MI455X (gfx1250) — matrix operand order repaired and checked by reading, not compiled, not run on hardware
//
#include <hip/hip_runtime.h>

// ---------------------------------------------------------------------------
// MI455X (gfx1250) multi-head attention forward.
// All matmuls via v_wmma_f32_16x16x32_bf16 (wave32 WMMA), fp32 accumulate.
// K-chunk staging in attention uses a double-buffered Tensor Data Mover
// pipeline when the builtin exists.
// ---------------------------------------------------------------------------

typedef unsigned short u16;
typedef unsigned long long u64;
typedef __attribute__((ext_vector_type(16))) __bf16 v16bf;   // WMMA A/B frag
typedef __attribute__((ext_vector_type(8)))  float  v8f;     // WMMA C/D frag
typedef __attribute__((ext_vector_type(4)))  unsigned int u32x4;
typedef __attribute__((ext_vector_type(4)))  int i32x4;
typedef __attribute__((ext_vector_type(8)))  int i32x8;

#if defined(__has_builtin)
#if __has_builtin(__builtin_amdgcn_tensor_load_to_lds) && \
    __has_builtin(__builtin_amdgcn_s_wait_tensorcnt)
#define HAVE_TDM 1
#endif
#endif
#ifndef HAVE_TDM
#define HAVE_TDM 0
#endif

#define STR2(x) #x
#define STR(x) STR2(x)
#if HAVE_TDM
#pragma message("CDNA5 probe: TDM builtin path ENABLED, clang major " STR(__clang_major__))
#else
#pragma message("CDNA5 probe: TDM builtin NOT available (manual staging), clang major " STR(__clang_major__))
#endif

union BFrag {
  v16bf v;
  u32x4 q[2];
};

__device__ __forceinline__ u16 f32_bf16(float f) {
  union { float f; unsigned u; } x; x.f = f;
  unsigned r = x.u + 0x7FFFu + ((x.u >> 16) & 1u);   // round-to-nearest-even
  return (u16)(r >> 16);
}

// --------------------------- f32 -> bf16 convert ---------------------------
__global__ void cvt_kernel(const float* __restrict__ src, u16* __restrict__ dst, int n8) {
  int i = blockIdx.x * blockDim.x + threadIdx.x;
  if (i >= n8) return;
  const float* s = src + (size_t)i * 8;
  u16 tmp[8];
#pragma unroll
  for (int j = 0; j < 8; ++j) tmp[j] = f32_bf16(s[j]);
  *(u32x4*)(dst + (size_t)i * 8) = *(const u32x4*)tmp;
}

// --------------------------- QKV projection GEMM ---------------------------
// Xb: [4096,1024] bf16, Wb: [3072,1024] bf16. 32x64 tile per wave, 8 WMMA per
// K-step, double-buffered fragment loads. Writes Q/K/V bf16 in [B,H,L,64];
// Q pre-scaled by 1/sqrt(64).
__global__ void qkv_gemm_kernel(const u16* __restrict__ Xb,
                                const u16* __restrict__ Wb,
                                u16* __restrict__ Qo,
                                u16* __restrict__ Ko,
                                u16* __restrict__ Vo) {
  const int lane = threadIdx.x & 31;
  const int lr = lane & 15;
  const int hl = (lane >> 4) & 1;
  const int mt = blockIdx.x & 127;     // 128 M tiles of 32 rows
  const int nt = blockIdx.x >> 7;      // 48 N groups of 64
  const int m0 = mt * 32;
  const int n0 = nt * 64;

  const u16* arow0 = Xb + (m0 + lr) * 1024 + hl * 8;         // A rows 0..15
  const u16* arow1 = arow0 + 16 * 1024;                      // A rows 16..31
  const u16* brow  = Wb + (n0 + lr) * 1024 + hl * 8;

  BFrag a[2][2], bf[2][4];
  auto loadk = [&](int buf, int kb) {
    a[buf][0].q[0] = *(const u32x4*)(arow0 + kb);
    a[buf][0].q[1] = *(const u32x4*)(arow0 + kb + 16);
    a[buf][1].q[0] = *(const u32x4*)(arow1 + kb);
    a[buf][1].q[1] = *(const u32x4*)(arow1 + kb + 16);
#pragma unroll
    for (int s = 0; s < 4; ++s) {
      const u16* p = brow + s * 16 * 1024 + kb;
      bf[buf][s].q[0] = *(const u32x4*)(p);
      bf[buf][s].q[1] = *(const u32x4*)(p + 16);
    }
  };

  v8f acc[2][4] = {};
  loadk(0, 0);
  for (int kb = 0; kb < 1024; kb += 32) {
    const int cur = (kb >> 5) & 1;
    __builtin_prefetch(arow0 + kb + 64, 0, 0);               // global_prefetch_b8
    loadk(cur ^ 1, (kb + 32) & 1023);                        // in-flight next step
#pragma unroll
    for (int mi = 0; mi < 2; ++mi)
#pragma unroll
      for (int s = 0; s < 4; ++s)
        acc[mi][s] = __builtin_amdgcn_wmma_f32_16x16x32_bf16(
            false, a[cur][mi].v, false, bf[cur][s].v, (short)0, acc[mi][s], false, false);
  }

  // n0 is 64-aligned: whole tile targets one of q/k/v, one head.
  const int which = n0 >> 10;                        // 0=q 1=k 2=v
  u16* dst = which == 0 ? Qo : (which == 1 ? Ko : Vo);
  const float scale = (which == 0) ? 0.125f : 1.0f;  // HEAD_DIM^-0.5
  const int h = (n0 & 1023) >> 6;
#pragma unroll
  for (int mi = 0; mi < 2; ++mi)
#pragma unroll
    for (int s = 0; s < 4; ++s)
#pragma unroll
      for (int r = 0; r < 8; ++r) {
        int row = m0 + mi * 16 + r + hl * 8;         // = b*2048 + l
        int d = s * 16 + lr;
        size_t off = ((((size_t)(row >> 11)) * 16 + h) * 2048 + (row & 2047)) * 64 + d;
        dst[off] = f32_bf16(acc[mi][s][r] * scale);
      }
}

// ------------------------------ flash attention ----------------------------
// 4 waves/block; each wave owns 16 queries; block streams 32-key chunks of
// K,V through LDS. K chunks ride a double-buffered TDM pipeline: wave 0
// issues chunk i+1, computes on chunk i, waits s_wait_tensorcnt<=1.
__global__ __launch_bounds__(128) void attn_kernel(const u16* __restrict__ Q,
                                                   const u16* __restrict__ K,
                                                   const u16* __restrict__ V,
                                                   u16* __restrict__ O) {
  __shared__ __align__(16) u16 Kch[2][32 * 64];    // [buf][key][d] row-major
  __shared__ __align__(16) u16 Vt[64 * 32];        // [d][key] transposed
  __shared__ __align__(16) u16 Pst[4][16 * 32];    // per-wave P staging

  const int lane = threadIdx.x & 31;
  const int w    = threadIdx.x >> 5;
  const int lr   = lane & 15;
  const int hl   = (lane >> 4) & 1;

  const int bh = blockIdx.x >> 5;                  // b*16+h
  const int qt = blockIdx.x & 31;
  const int b  = bh >> 4;
  const int h  = bh & 15;
  const int q0 = qt * 64 + w * 16;

  // Q fragments (A-layout), already scaled by 1/sqrt(d)
  BFrag aQ[2];
  {
    const u16* qb = Q + (((size_t)bh * 2048) + q0 + lr) * 64;
#pragma unroll
    for (int dc = 0; dc < 2; ++dc) {
      const u16* r = qb + dc * 32 + hl * 8;
      aQ[dc].q[0] = *(const u32x4*)(r);
      aQ[dc].q[1] = *(const u32x4*)(r + 16);
    }
  }

  float m[8], lsum[8];
  v8f Oacc[4] = {};
#pragma unroll
  for (int r = 0; r < 8; ++r) { m[r] = -1e30f; lsum[r] = 0.f; }

  const u16* kbase = K + ((size_t)bh * 2048) * 64;
  const u16* vbase = V + ((size_t)bh * 2048) * 64;

#if HAVE_TDM
  // Constant D# pieces: 2D tensor, data_size=2B, tile 64x32, stride 64.
  i32x8 g1 = {};
  g1[0] = (1 << 16);               // workgroup_mask=0, data_size=1 (2 bytes)
  g1[1] = (int)(64u << 16);        // tensor_dim0 = 64 (low 16 in [63:48])
  g1[2] = (int)(2048u << 16);      // dim0 hi=0, tensor_dim1 = 2048 (low 16)
  g1[3] = (int)(64u << 16);        // dim1 hi=0, tile_dim0 = 64
  g1[4] = 32;                      // tile_dim1 = 32, tile_dim2 = 0
  g1[5] = 64;                      // tensor_dim0_stride = 64
  g1[6] = 0;
  g1[7] = 0;
  const i32x4 g2 = {};             // dims 2..3 unused (2D tensor)
  const i32x4 g3 = {};
#if __clang_major__ >= 23
  const i32x8 g4 = {};
#endif
  const unsigned lds_k0 = (unsigned)(size_t)(void*)Kch[0];
  const unsigned lds_k1 = (unsigned)(size_t)(void*)Kch[1];
  const u64 kbytes = (u64)(size_t)kbase;

  auto issue_k = [&](unsigned ldsaddr, int kcc) {
    u64 ga = kbytes + (u64)kcc * 128;               // 32 keys * 64 d * 2B
    u32x4 g0;
    g0[0] = 1u;                                     // count=1, user descriptor
    g0[1] = ldsaddr;                                // LDS destination
    g0[2] = (unsigned)(ga & 0xffffffffu);
    g0[3] = (unsigned)((ga >> 32) & 0x01ffffffu) | (2u << 30);  // type=2
#if __clang_major__ >= 23
    __builtin_amdgcn_tensor_load_to_lds(g0, g1, g2, g3, g4, 0);
#else
    __builtin_amdgcn_tensor_load_to_lds(g0, g1, g2, g3, 0);
#endif
  };
  if (w == 0) issue_k(lds_k0, 0);                   // prologue: chunk 0 in flight
#endif

  for (int kc = 0; kc < 2048; kc += 32) {
    const int cur = (kc >> 5) & 1;
    __syncthreads();                                // iter i-1 reads complete
#if HAVE_TDM
    const bool more = (kc + 32) < 2048;
    if (w == 0 && more)                             // overlap next DMA with compute
      issue_k(cur ? lds_k0 : lds_k1, kc + 32);
#else
    {   // manual K staging, contiguous 4KB, 256 x b128
      const u32x4* src = (const u32x4*)(kbase + (size_t)kc * 64);
      u32x4* dst = (u32x4*)Kch[cur];
      dst[threadIdx.x]       = src[threadIdx.x];
      dst[threadIdx.x + 128] = src[threadIdx.x + 128];
    }
#endif
    {   // stage V chunk transposed: Vt[d][key]  (overlaps the TDM transfer)
      const u16* src = vbase + (size_t)kc * 64;
#pragma unroll
      for (int t = 0; t < 2; ++t) {
        int idx = threadIdx.x + t * 128;
        int key = idx >> 3;
        int d0 = (idx & 7) * 8;
        u32x4 pkt = *(const u32x4*)(src + key * 64 + d0);
        const u16* e = (const u16*)&pkt;
#pragma unroll
        for (int j = 0; j < 8; ++j) Vt[(d0 + j) * 32 + key] = e[j];
      }
    }
#if HAVE_TDM
    if (w == 0) {                                   // in-order per-wave TDM:
      if (more) __builtin_amdgcn_s_wait_tensorcnt(1);   // chunk i landed
      else      __builtin_amdgcn_s_wait_tensorcnt(0);   // last chunk landed
    }
#endif
    __syncthreads();
    const u16* Kc = Kch[cur];

    // S = Q K^T : two 16-key tiles, 2 WMMAs each (d split 0..31 / 32..63)
    v8f sC[2];
#pragma unroll
    for (int t = 0; t < 2; ++t) {
      v8f s = {};
#pragma unroll
      for (int dc = 0; dc < 2; ++dc) {
        BFrag bK;  // B[K=d, N=key]: lane holds key, d contiguous
        const u16* r = &Kc[(t * 16 + lr) * 64 + dc * 32 + hl * 8];
        bK.q[0] = *(const u32x4*)(r);
        bK.q[1] = *(const u32x4*)(r + 16);
        s = __builtin_amdgcn_wmma_f32_16x16x32_bf16(false, aQ[dc].v, false, bK.v,
                                                    (short)0, s, false, false);
      }
      sC[t] = s;
    }

    // Online softmax: row r lives across the lane-half in VGPR r of the C frag
    float fac[8];
#pragma unroll
    for (int r = 0; r < 8; ++r) {
      float tmax = fmaxf(sC[0][r], sC[1][r]);
      tmax = fmaxf(tmax, __shfl_xor(tmax, 1, 32));
      tmax = fmaxf(tmax, __shfl_xor(tmax, 2, 32));
      tmax = fmaxf(tmax, __shfl_xor(tmax, 4, 32));
      tmax = fmaxf(tmax, __shfl_xor(tmax, 8, 32));
      float mn = fmaxf(m[r], tmax);
      float f = __expf(m[r] - mn);
      float p0 = __expf(sC[0][r] - mn);
      float p1 = __expf(sC[1][r] - mn);
      sC[0][r] = p0; sC[1][r] = p1;
      float rs = p0 + p1;
      rs += __shfl_xor(rs, 1, 32);
      rs += __shfl_xor(rs, 2, 32);
      rs += __shfl_xor(rs, 4, 32);
      rs += __shfl_xor(rs, 8, 32);
      lsum[r] = lsum[r] * f + rs;
      m[r] = mn;
      fac[r] = f;
    }
#pragma unroll
    for (int dt = 0; dt < 4; ++dt)
#pragma unroll
      for (int r = 0; r < 8; ++r) Oacc[dt][r] *= fac[r];

    // C-layout -> A-layout via wave-private LDS staging (DScnt keeps order)
    u16* P = Pst[w];
#pragma unroll
    for (int t = 0; t < 2; ++t)
#pragma unroll
      for (int r = 0; r < 8; ++r)
        P[(r + hl * 8) * 32 + t * 16 + lr] = f32_bf16(sC[t][r]);

    BFrag aP;
    {
      const u16* pr = &P[lr * 32 + hl * 8];
      aP.q[0] = *(const u32x4*)(pr);
      aP.q[1] = *(const u32x4*)(pr + 16);
    }
#pragma unroll
    for (int dt = 0; dt < 4; ++dt) {
      BFrag bV;  // B[K=key, N=d]: lane holds d, key contiguous in Vt
      const u16* vr = &Vt[(dt * 16 + lr) * 32 + hl * 8];
      bV.q[0] = *(const u32x4*)(vr);
      bV.q[1] = *(const u32x4*)(vr + 16);
      Oacc[dt] = __builtin_amdgcn_wmma_f32_16x16x32_bf16(false, aP.v, false, bV.v,
                                                         (short)0, Oacc[dt], false, false);
    }
  }

  // Normalize and store bf16 O in [B, L, HIDDEN] (c = h*64 + d)
#pragma unroll
  for (int r = 0; r < 8; ++r) lsum[r] = 1.0f / lsum[r];
#pragma unroll
  for (int dt = 0; dt < 4; ++dt)
#pragma unroll
    for (int r = 0; r < 8; ++r) {
      int row = q0 + r + hl * 8;
      size_t off = (((size_t)b * 2048) + row) * 1024 + h * 64 + dt * 16 + lr;
      O[off] = f32_bf16(Oacc[dt][r] * lsum[r]);
    }
}

// ------------------------- output projection GEMM --------------------------
// out[b,o,l] = sum_c w_out[o,c] * O[b,l,c] + bias[o]; fp32 output [B,1024,2048]
__global__ void out_gemm_kernel(const u16* __restrict__ Wb,     // [1024,1024] bf16
                                const u16* __restrict__ Ob,     // [B,L,1024] bf16
                                const float* __restrict__ bias,
                                float* __restrict__ out) {
  const int lane = threadIdx.x & 31;
  const int lr = lane & 15;
  const int hl = (lane >> 4) & 1;
  int t = blockIdx.x;
  const int b = t >> 10;
  const int mt = t & 31;               // 32 o-tiles of 32
  const int nt = (t >> 5) & 31;        // 32 l-groups of 64
  const int m0 = mt * 32;
  const int n0 = nt * 64;

  const u16* arow0 = Wb + (m0 + lr) * 1024 + hl * 8;
  const u16* arow1 = arow0 + 16 * 1024;
  const u16* brow  = Ob + (((size_t)b * 2048) + n0 + lr) * 1024 + hl * 8;

  BFrag a[2][2], bf[2][4];
  auto loadk = [&](int buf, int kb) {
    a[buf][0].q[0] = *(const u32x4*)(arow0 + kb);
    a[buf][0].q[1] = *(const u32x4*)(arow0 + kb + 16);
    a[buf][1].q[0] = *(const u32x4*)(arow1 + kb);
    a[buf][1].q[1] = *(const u32x4*)(arow1 + kb + 16);
#pragma unroll
    for (int s = 0; s < 4; ++s) {
      const u16* p = brow + s * 16 * 1024 + kb;
      bf[buf][s].q[0] = *(const u32x4*)(p);
      bf[buf][s].q[1] = *(const u32x4*)(p + 16);
    }
  };

  v8f acc[2][4] = {};
  loadk(0, 0);
  for (int kb = 0; kb < 1024; kb += 32) {
    const int cur = (kb >> 5) & 1;
    __builtin_prefetch(arow0 + kb + 64, 0, 0);
    loadk(cur ^ 1, (kb + 32) & 1023);
#pragma unroll
    for (int mi = 0; mi < 2; ++mi)
#pragma unroll
      for (int s = 0; s < 4; ++s)
        acc[mi][s] = __builtin_amdgcn_wmma_f32_16x16x32_bf16(
            false, a[cur][mi].v, false, bf[cur][s].v, (short)0, acc[mi][s], false, false);
  }

  float bv[2][8];
#pragma unroll
  for (int mi = 0; mi < 2; ++mi)
#pragma unroll
    for (int r = 0; r < 8; ++r) bv[mi][r] = bias[m0 + mi * 16 + r + hl * 8];
#pragma unroll
  for (int mi = 0; mi < 2; ++mi)
#pragma unroll
    for (int s = 0; s < 4; ++s)
#pragma unroll
      for (int r = 0; r < 8; ++r) {
        int orow = m0 + mi * 16 + r + hl * 8;
        int lcol = n0 + s * 16 + lr;
        out[(((size_t)b * 1024) + orow) * 2048 + lcol] = acc[mi][s][r] + bv[mi][r];
      }
}

// --------------------------------- launch ----------------------------------
extern "C" void kernel_launch(void* const* d_in, const int* in_sizes, int n_in,
                              void* d_out, int out_size, void* d_ws, size_t ws_size,
                              hipStream_t stream) {
  const float* x     = (const float*)d_in[0];   // [2,2048,1024]
  const float* w_qkv = (const float*)d_in[1];   // [3072,1024]
  const float* w_out = (const float*)d_in[2];   // [1024,1024]
  const float* b_out = (const float*)d_in[3];   // [1024]
  float* out = (float*)d_out;                   // [2,1024,2048]

  char* ws = (char*)d_ws;                       // 48 MB carve-up
  u16* xb  = (u16*)(ws);                        //  0 MB: x bf16     (8 MB)
  u16* wqb = (u16*)(ws + ((size_t)8  << 20));   //  8 MB: w_qkv bf16 (6 MB)
  u16* wob = (u16*)(ws + ((size_t)14 << 20));   // 14 MB: w_out bf16 (2 MB)
  u16* qws = (u16*)(ws + ((size_t)16 << 20));   // 16 MB: Q bf16     (8 MB)
  u16* kws = (u16*)(ws + ((size_t)24 << 20));   // 24 MB: K bf16     (8 MB)
  u16* vws = (u16*)(ws + ((size_t)32 << 20));   // 32 MB: V bf16     (8 MB)
  u16* ows = (u16*)(ws + ((size_t)40 << 20));   // 40 MB: O bf16     (8 MB)

  cvt_kernel<<<(524288 + 255) / 256, 256, 0, stream>>>(x, xb, 524288);      // 4M/8
  cvt_kernel<<<(393216 + 255) / 256, 256, 0, stream>>>(w_qkv, wqb, 393216); // 3M/8
  cvt_kernel<<<(131072 + 255) / 256, 256, 0, stream>>>(w_out, wob, 131072); // 1M/8

  qkv_gemm_kernel<<<128 * 48, 32, 0, stream>>>(xb, wqb, qws, kws, vws);
  attn_kernel<<<1024, 128, 0, stream>>>(qws, kws, vws, ows);
  out_gemm_kernel<<<2048, 32, 0, stream>>>(wob, ows, b_out, out);
}
